// SHCSATritonSoftmax_55860344652356
// MI455X (gfx1250) — compile-verified, matrix operand order checked by reading, not run on hardware
//
#include <hip/hip_runtime.h>

// MI455X / gfx1250 fused causal attention:
//   K1: qkv = x@W+b (f16 WMMA, f32 accum); W tiles double-buffered via async
//       global->LDS DMA (ASYNCcnt), pipelined one K-step ahead
//   K2: flash attention, 16 q-rows/block; k and v blocks double-buffered via
//       async global->LDS DMA, pipelined one key-block ahead; online softmax
// Workspace: 3 * 4096*1024 * 2B = 24 MiB.

typedef __attribute__((ext_vector_type(16))) _Float16 v16h;
typedef __attribute__((ext_vector_type(8)))  float    v8f;

#define T_DIM 4096
#define C_DIM 1024
#define N3    3072

union FragH { v16h v; unsigned u[8]; _Float16 h[16]; };
union Pk2  { unsigned u; _Float16 h[2]; };

// ISA 7.12.2: 16-bit A/B fragment, kh=lane>>4 selects K-half.
// VGPR v holds K pair {kb, kb+1}:
//   v=0..3 -> K = kh*8 + 2v ; v=4..7 -> K = 16 + kh*8 + 2(v-4)
__device__ __forceinline__ int kbase(int v, int kh) {
  return ((v < 4) ? 0 : 16) + kh * 8 + 2 * (v & 3);
}

static __device__ __forceinline__ v8f wmma_f16(v16h a, v16h b, v8f c) {
  return __builtin_amdgcn_wmma_f32_16x16x32_f16(false, a, false, b,
                                                (short)0, c, false, false);
}

// gfx1250 async DMA: global -> LDS, 16B per active lane, tracked by ASYNCcnt.
__device__ __forceinline__ void async_b128(unsigned lds_byte_off, const void* gptr) {
  asm volatile("global_load_async_to_lds_b128 %0, %1, off"
               :: "v"(lds_byte_off), "v"(gptr) : "memory");
}
// Async loads complete in order: waiting <=N drains the older group while the
// most recently issued N stay in flight (pipelined double-buffering).
__device__ __forceinline__ void wait_async_le0()  { asm volatile("s_wait_asynccnt 0x0"  ::: "memory"); }
__device__ __forceinline__ void wait_async_le4()  { asm volatile("s_wait_asynccnt 0x4"  ::: "memory"); }
__device__ __forceinline__ void wait_async_le32() { asm volatile("s_wait_asynccnt 0x20" ::: "memory"); }
// Generic pointer to LDS: hardware LDS address = addr[31:0] (ISA 10.2 aperture).
__device__ __forceinline__ unsigned lds_off(const void* p) {
  return (unsigned)(unsigned long long)p;
}

// ---------------------------------------------------------------------------
// Kernel 1: qkv projection. Grid (24, 32), 256 threads (8 waves).
// Wave (wm,wn) computes a 64x32 sub-tile of the 128x128 block as 4x2 WMMA
// tiles. W tiles [32K x 128N] f32 double-buffered in LDS via async DMA.
// ---------------------------------------------------------------------------
__global__ __launch_bounds__(256) void qkv_gemm(
    const float* __restrict__ x, const float* __restrict__ W,
    const float* __restrict__ bias,
    _Float16* __restrict__ qs, _Float16* __restrict__ kk,
    _Float16* __restrict__ vT)
{
  __shared__ __align__(16) float wf[2][32 * 128];   // 2 x 16 KB staged W tiles

  const int lane = threadIdx.x & 31;
  const int wv   = threadIdx.x >> 5;   // 0..7
  const int wm   = wv & 1;             // M half (64 rows)
  const int wn   = wv >> 1;            // N quarter (32 cols)
  const int ln   = lane & 15;
  const int kh   = lane >> 4;

  const int rowb = blockIdx.y * 128 + wm * 64;
  const int cb   = blockIdx.x * 128;            // block column base (uniform)
  const int colb = cb + wn * 32;

  auto stage_w = [&](int k0s, int bsel) {
    #pragma unroll
    for (int c = 0; c < 4; ++c) {
      const int idx = c * 256 + threadIdx.x;    // 0..1023 chunks of 4 floats
      const int kr  = idx >> 5;                 // 0..31
      const int co  = (idx & 31) * 4;           // 0..124
      async_b128(lds_off(&wf[bsel][kr * 128 + co]),
                 W + (size_t)(k0s + kr) * N3 + cb + co);
    }
  };

  v8f acc[4][2];
  #pragma unroll
  for (int i = 0; i < 4; ++i)
    #pragma unroll
    for (int j = 0; j < 2; ++j) acc[i][j] = {};

  stage_w(0, 0);                                // prologue: fill buffer 0
  int buf = 0;
  for (int k0 = 0; k0 < C_DIM; k0 += 32, buf ^= 1) {
    // ---- A fragments from global (K-contiguous, vectorizes to b128)
    FragH a[4];
    #pragma unroll
    for (int mt = 0; mt < 4; ++mt) {
      const float* xp = x + (size_t)(rowb + mt * 16 + ln) * C_DIM + k0;
      #pragma unroll
      for (int v = 0; v < 8; ++v) {
        float2 xv = *(const float2*)(xp + kbase(v, kh));
        a[mt].h[2 * v]     = (_Float16)xv.x;
        a[mt].h[2 * v + 1] = (_Float16)xv.y;
      }
    }

    const bool more = (k0 + 32) < C_DIM;
    if (more) {                                 // pipeline next W tile
      stage_w(k0 + 32, buf ^ 1);
      wait_async_le4();                         // drain only the older group
    } else {
      wait_async_le0();
    }
    __syncthreads();                            // wf[buf] visible to all waves

    // ---- B fragments from LDS (lanes read consecutive columns)
    FragH b[2];
    #pragma unroll
    for (int nt = 0; nt < 2; ++nt) {
      const int cl = wn * 32 + nt * 16 + ln;    // column within block
      #pragma unroll
      for (int v = 0; v < 8; ++v) {
        const int K = kbase(v, kh);
        b[nt].h[2 * v]     = (_Float16)wf[buf][K * 128 + cl];
        b[nt].h[2 * v + 1] = (_Float16)wf[buf][(K + 1) * 128 + cl];
      }
    }
    #pragma unroll
    for (int mt = 0; mt < 4; ++mt)
      #pragma unroll
      for (int nt = 0; nt < 2; ++nt)
        acc[mt][nt] = wmma_f16(a[mt].v, b[nt].v, acc[mt][nt]);
    __syncthreads();                            // readers done before restage
  }

  // Epilogue: bias + route. The 128-col block lies entirely inside one of
  // q / k / v (1024 % 128 == 0), so the region branch is block-uniform.
  const int region = cb >> 10;                  // 0:q  1:k  2:v
  if (region == 0) {
    #pragma unroll
    for (int mt = 0; mt < 4; ++mt)
      #pragma unroll
      for (int nt = 0; nt < 2; ++nt) {
        const int col = colb + nt * 16 + ln;
        const float bi = bias[col];
        #pragma unroll
        for (int r = 0; r < 8; ++r) {
          const int row = rowb + mt * 16 + r + 8 * kh;
          qs[(size_t)row * C_DIM + col] =
              (_Float16)((acc[mt][nt][r] + bi) * 0.03125f);
        }
      }
  } else if (region == 1) {
    #pragma unroll
    for (int mt = 0; mt < 4; ++mt)
      #pragma unroll
      for (int nt = 0; nt < 2; ++nt) {
        const int col = colb + nt * 16 + ln;
        const float bi = bias[col];
        #pragma unroll
        for (int r = 0; r < 8; ++r) {
          const int row = rowb + mt * 16 + r + 8 * kh;
          kk[(size_t)row * C_DIM + (col - 1024)] =
              (_Float16)(acc[mt][nt][r] + bi);
        }
      }
  } else {
    // v^T: adjacent accumulator rows are adjacent in memory -> packed b32.
    #pragma unroll
    for (int mt = 0; mt < 4; ++mt)
      #pragma unroll
      for (int nt = 0; nt < 2; ++nt) {
        const int col = colb + nt * 16 + ln;
        const float bi = bias[col];
        const size_t vrow = (size_t)(col - 2048) * T_DIM;
        #pragma unroll
        for (int r = 0; r < 8; r += 2) {
          const int row = rowb + mt * 16 + r + 8 * kh;   // even
          Pk2 p;
          p.h[0] = (_Float16)(acc[mt][nt][r] + bi);
          p.h[1] = (_Float16)(acc[mt][nt][r + 1] + bi);
          *(unsigned*)(vT + vrow + row) = p.u;
        }
      }
  }
}

// ---------------------------------------------------------------------------
// Kernel 2: flash attention. Grid (256): block qb handles q rows 16qb..16qb+15.
// 8 waves; wave wv computes score partials over features [128wv,128wv+128) and
// owns output features [128wv,128wv+128) as 8 WMMA accumulator tiles.
// Key blocks of Bc=32; k block [32x1024] and v block [1024x32] (from v^T) are
// double-buffered in 256 KiB dynamic LDS via pipelined async DMA.
// ---------------------------------------------------------------------------
__global__ __launch_bounds__(256) void flash_attn(
    const _Float16* __restrict__ qs, const _Float16* __restrict__ kk,
    const _Float16* __restrict__ vT, float* __restrict__ out)
{
  extern __shared__ _Float16 dynls[];
  // layout: ks[2][32][1024] then vs[2][1024][32]
  _Float16* ks0 = dynls;
  _Float16* vs0 = dynls + 2 * 32 * 1024;

  __shared__ float    sp[8][16][32];          // per-wave score partials
  __shared__ float    ssum[16][32];           // reduced + masked scores
  __shared__ _Float16 pbuf[16][32];           // P = exp(S - m) in f16
  __shared__ float    alpha_l[16];            // per-row rescale factors
  __shared__ float    lfin[16];               // 1/l per row

  const int qb   = blockIdx.x;
  const int R0   = qb * 16;
  const int lane = threadIdx.x & 31;
  const int wv   = threadIdx.x >> 5;
  const int ln   = lane & 15;
  const int kh   = lane >> 4;
  const int fb   = wv * 128;                  // this wave's feature-slice base

  auto stage_kv = [&](int jj, int bsel) {
    const int c0 = jj * 32;
    _Float16* ksb = ks0 + bsel * (32 * 1024);
    _Float16* vsb = vs0 + bsel * (32 * 1024);
    #pragma unroll
    for (int c = 0; c < 16; ++c) {
      const int idx = c * 256 + threadIdx.x;  // 0..4095 chunks of 8 halves
      const int key = idx >> 7;               // 0..31
      const int fo  = (idx & 127) * 8;        // 0..1016
      async_b128(lds_off(ksb + key * C_DIM + fo),
                 kk + (size_t)(c0 + key) * C_DIM + fo);
    }
    #pragma unroll
    for (int c = 0; c < 16; ++c) {
      const int idx = c * 256 + threadIdx.x;  // 0..4095
      const int ft  = idx >> 2;               // 0..1023
      const int kc  = (idx & 3) * 8;          // 0,8,16,24
      async_b128(lds_off(vsb + ft * 32 + kc),
                 vT + (size_t)ft * T_DIM + c0 + kc);
    }
  };

  // q fragments for this wave's K-slice: persistent across the j loop.
  FragH qa[4];
  {
    const _Float16* qp = qs + (size_t)(R0 + ln) * C_DIM + fb;
    #pragma unroll
    for (int f = 0; f < 4; ++f)
      #pragma unroll
      for (int v = 0; v < 8; ++v)
        qa[f].u[v] = *(const unsigned*)(qp + f * 32 + kbase(v, kh));
  }

  v8f acc[8];
  #pragma unroll
  for (int t = 0; t < 8; ++t) acc[t] = {};

  float m_run = -__builtin_inff();            // live in wave 0, lanes 0..15
  float l_run = 0.0f;

  const int jend = (R0 + 15) >> 5;
  stage_kv(0, 0);                             // prologue: fill buffer 0
  for (int j = 0; j <= jend; ++j) {
    const int  bsel = j & 1;
    const int  col0 = j * 32;
    const bool more = j < jend;
    const _Float16* ksb = ks0 + bsel * (32 * 1024);
    const _Float16* vsb = vs0 + bsel * (32 * 1024);

    if (more) {                               // pipeline next k/v block
      stage_kv(j + 1, bsel ^ 1);
      wait_async_le32();                      // drain only the older group
    } else {
      wait_async_le0();
    }
    if (col0 + 64 < T_DIM)                    // warm L2 two blocks ahead
      __builtin_prefetch(kk + (size_t)(col0 + 64 + lane) * C_DIM + fb, 0, 3);
    __syncthreads();                          // staged block visible

    // ---- partial scores S[16x32] over this wave's 128-feature slice
    v8f s0 = {}, s1 = {};
    #pragma unroll
    for (int f = 0; f < 4; ++f) {
      FragH b0, b1;
      #pragma unroll
      for (int v = 0; v < 8; ++v) {
        const int K = fb + 32 * f + kbase(v, kh);
        b0.u[v] = *(const unsigned*)(ksb + (size_t)ln * C_DIM + K);
        b1.u[v] = *(const unsigned*)(ksb + (size_t)(16 + ln) * C_DIM + K);
      }
      s0 = wmma_f16(qa[f].v, b0.v, s0);
      s1 = wmma_f16(qa[f].v, b1.v, s1);
    }

    #pragma unroll
    for (int r = 0; r < 8; ++r) {
      sp[wv][r + 8 * kh][ln]      = s0[r];
      sp[wv][r + 8 * kh][16 + ln] = s1[r];
    }
    __syncthreads();

    // ---- reduce partials across waves, apply causal mask (col <= row)
    {
      const int tid = threadIdx.x;
      #pragma unroll
      for (int e0 = 0; e0 < 2; ++e0) {
        const int e = tid + e0 * 256;
        const int m = e >> 5, c = e & 31;
        float s = 0.f;
        #pragma unroll
        for (int w2 = 0; w2 < 8; ++w2) s += sp[w2][m][c];
        ssum[m][c] = (col0 + c <= R0 + m) ? s : -__builtin_inff();
      }
    }
    __syncthreads();

    // ---- wave 0: online softmax update (one lane per row)
    if (wv == 0 && lane < 16) {
      const int row = lane;
      float mx = -__builtin_inff();
      for (int c = 0; c < 32; ++c) mx = fmaxf(mx, ssum[row][c]);
      const float mnew = fmaxf(m_run, mx);    // finite: diagonal always valid
      const float al   = __expf(m_run - mnew);
      float ps = 0.f;
      for (int c = 0; c < 32; ++c) {
        const float pv = __expf(ssum[row][c] - mnew);
        pbuf[row][c] = (_Float16)pv;
        ps += pv;
      }
      l_run = al * l_run + ps;
      m_run = mnew;
      alpha_l[row] = al;
    }
    __syncthreads();

    // ---- rescale accumulators by alpha[row]
    float alv[8];
    #pragma unroll
    for (int r = 0; r < 8; ++r) alv[r] = alpha_l[r + 8 * kh];
    #pragma unroll
    for (int t = 0; t < 8; ++t)
      #pragma unroll
      for (int r = 0; r < 8; ++r) acc[t][r] *= alv[r];

    // ---- acc += P @ V  (both operands from LDS, K-contiguous)
    FragH pa;
    #pragma unroll
    for (int v = 0; v < 8; ++v)
      pa.u[v] = *(const unsigned*)(&pbuf[ln][kbase(v, kh)]);

    #pragma unroll
    for (int t = 0; t < 8; ++t) {
      FragH bv;
      #pragma unroll
      for (int v = 0; v < 8; ++v)
        bv.u[v] = *(const unsigned*)(vsb + (size_t)(fb + t * 16 + ln) * 32 +
                                     kbase(v, kh));
      acc[t] = wmma_f16(pa.v, bv.v, acc[t]);
    }
    __syncthreads();   // all reads of this buffer done before it is restaged
  }

  if (wv == 0 && lane < 16) lfin[lane] = 1.0f / l_run;
  __syncthreads();

  float linv[8];
  #pragma unroll
  for (int r = 0; r < 8; ++r) linv[r] = lfin[r + 8 * kh];
  #pragma unroll
  for (int t = 0; t < 8; ++t) {
    const int feat = fb + t * 16 + ln;
    #pragma unroll
    for (int r = 0; r < 8; ++r)
      out[(size_t)(R0 + r + 8 * kh) * C_DIM + feat] = acc[t][r] * linv[r];
  }
}

// ---------------------------------------------------------------------------
extern "C" void kernel_launch(void* const* d_in, const int* in_sizes, int n_in,
                              void* d_out, int out_size, void* d_ws, size_t ws_size,
                              hipStream_t stream) {
  const float* x    = (const float*)d_in[0];   // [4096,1024]
  const float* W    = (const float*)d_in[1];   // [1024,3072]
  const float* bias = (const float*)d_in[2];   // [3072]
  float* out = (float*)d_out;                  // [4096,1024]

  _Float16* qs = (_Float16*)d_ws;                     // [4096][1024] scaled q
  _Float16* kk = qs + (size_t)T_DIM * C_DIM;          // [4096][1024]
  _Float16* vT = kk + (size_t)T_DIM * C_DIM;          // [1024][4096] v^T
  (void)in_sizes; (void)n_in; (void)out_size; (void)ws_size;

  dim3 g1(N3 / 128, T_DIM / 128), b1(256);
  qkv_gemm<<<g1, b1, 0, stream>>>(x, W, bias, qs, kk, vT);

  dim3 g2(T_DIM / 16), b2(256);
  flash_attn<<<g2, b2, 256 * 1024, stream>>>(qs, kk, vT, out);
}
